// Block_2018634629560
// MI455X (gfx1250) — compile-verified, matrix operand order checked by reading, not run on hardware
//
#include <hip/hip_runtime.h>

// ---------------------------------------------------------------------------
// MI455X (gfx1250, wave32) implementation of the attention + gated-DeltaNet
// block.  All large GEMMs use v_wmma_f32_16x16x32_bf16; attention uses a
// flash-style WMMA kernel with TDM (tensor_load_to_lds) staging of K/V tiles;
// the sequential DeltaNet scan keeps its 128x128 state in VGPRs.
// ---------------------------------------------------------------------------

typedef unsigned short u16;
typedef __bf16 bf16_t;
typedef bf16_t v16bf __attribute__((ext_vector_type(16)));
typedef float  v8f   __attribute__((ext_vector_type(8)));
typedef unsigned int v4u __attribute__((ext_vector_type(4)));
typedef int v8i __attribute__((ext_vector_type(8)));
typedef int v4i __attribute__((ext_vector_type(4)));

#define T_SEQ 2048
#define D_MODEL 2048

__device__ __forceinline__ u16 f32_to_bf16(float f) {
    unsigned int u = __float_as_uint(f);
    unsigned int r = (u + 0x7FFFu + ((u >> 16) & 1u)) >> 16;
    return (u16)r;
}

// ---------------------------------------------------------------------------
// WMMA fragment loader (CDNA5 16-bit A layout; B of row-major [N,K] is the
// mirror image, so the same loader serves both).
// ---------------------------------------------------------------------------
union FragU { uint4 q[2]; v16bf v; };

__device__ __forceinline__ v16bf load_frag_rowmajor(const u16* __restrict__ base,
                                                    long stride, int row0, int k0,
                                                    int lane) {
    int r  = row0 + (lane & 15);
    int hi = lane >> 4;
    const u16* p = base + (long)r * stride + k0 + 8 * hi;
    FragU f;
    f.q[0] = *(const uint4*)(p);
    f.q[1] = *(const uint4*)(p + 16);
    return f.v;
}

// ---------------------------------------------------------------------------
// Tensor Data Mover: DMA a 2D bf16 tile (tile_d0 elems/row, tile_d1 rows,
// row stride stride0 elems) from global into LDS at lds_off.
// D# layout per CDNA5 ISA 8.3/8.4 (group0: count/lds/global/type,
// group1: data_size + dims + strides).  Groups 2/3 zero (<=2D tensor).
// This toolchain's builtin takes 6 args (extra int32x8 before cpol).
// ---------------------------------------------------------------------------
__device__ __forceinline__ void tdm_load_2d_bf16(unsigned lds_off, const void* gptr,
                                                 unsigned tile_d0, unsigned tile_d1,
                                                 unsigned long long stride0) {
    unsigned long long ga = (unsigned long long)gptr;
    v4u g0;
    g0[0] = 1u;                                            // count=1 (user D#)
    g0[1] = lds_off;                                       // lds_addr (bytes)
    g0[2] = (unsigned)(ga & 0xFFFFFFFFu);                  // global_addr[31:0]
    g0[3] = (unsigned)((ga >> 32) & 0x01FFFFFFu)           // global_addr[56:32]
            | (2u << 30);                                  // type=2 ("image")
    v8i g1;
    g1[0] = (int)(1u << 16);                               // data_size=1 -> 2 bytes
    g1[1] = (int)((tile_d0 & 0xFFFFu) << 16);              // tensor_dim0[15:0]
    g1[2] = (int)(((tile_d0 >> 16) & 0xFFFFu)              // tensor_dim0[31:16]
            | ((tile_d1 & 0xFFFFu) << 16));                // tensor_dim1[15:0]
    g1[3] = (int)(((tile_d1 >> 16) & 0xFFFFu)              // tensor_dim1[31:16]
            | ((tile_d0 & 0xFFFFu) << 16));                // tile_dim0
    g1[4] = (int)(tile_d1 & 0xFFFFu);                      // tile_dim1 (tile_dim2=0)
    g1[5] = (int)(unsigned)(stride0 & 0xFFFFFFFFull);      // dim0_stride[31:0]
    g1[6] = (int)(unsigned)((stride0 >> 32) & 0xFFFFull);  // dim0_stride[47:32]
    g1[7] = 0;
    v4i gz4 = {0, 0, 0, 0};
    v8i gz8 = {0, 0, 0, 0, 0, 0, 0, 0};
    __builtin_amdgcn_tensor_load_to_lds(g0, g1, gz4, gz4, gz8, 0);
}

// ---------------------------------------------------------------------------
// GEMM: C[M,N] (f32) = A[M,K](bf16) @ W[N,K](bf16)^T  (+ optional residual)
// One wave computes a 64x32 output tile (4x2 sub-tiles of 16x16).
// ---------------------------------------------------------------------------
__global__ void __launch_bounds__(256) wmma_gemm_kernel(
    const u16* __restrict__ A, const u16* __restrict__ W,
    float* __restrict__ C, const float* __restrict__ residual,
    int M, int N, int K)
{
    int lane = threadIdx.x & 31;
    int wave = threadIdx.x >> 5;
    int tilesN = N >> 5;
    long tile   = (long)blockIdx.x * 8 + wave;
    long nTiles = (long)(M >> 6) * tilesN;
    if (tile >= nTiles) return;
    int m0 = (int)(tile / tilesN) << 6;
    int n0 = (int)(tile % tilesN) << 5;

    v8f acc[4][2];
#pragma unroll
    for (int i = 0; i < 4; i++)
#pragma unroll
        for (int j = 0; j < 2; j++)
#pragma unroll
            for (int v = 0; v < 8; v++) acc[i][j][v] = 0.0f;

    for (int k0 = 0; k0 < K; k0 += 32) {
        if (k0 + 64 <= K) {  // prefetch next k-slab -> global_prefetch_b8
            __builtin_prefetch(A + (long)(m0 + (lane & 15)) * K + k0 + 32, 0, 0);
            __builtin_prefetch(W + (long)(n0 + (lane & 15)) * K + k0 + 32, 0, 0);
        }
        v16bf a[4], b[2];
#pragma unroll
        for (int i = 0; i < 4; i++)
            a[i] = load_frag_rowmajor(A, K, m0 + 16 * i, k0, lane);
#pragma unroll
        for (int j = 0; j < 2; j++)
            b[j] = load_frag_rowmajor(W, K, n0 + 16 * j, k0, lane);
#pragma unroll
        for (int i = 0; i < 4; i++)
#pragma unroll
            for (int j = 0; j < 2; j++)
                acc[i][j] = __builtin_amdgcn_wmma_f32_16x16x32_bf16(
                    false, a[i], false, b[j], (short)0, acc[i][j], false, false);
    }

    int nc = n0 + (lane & 15);
    int mb = m0 + ((lane >> 4) << 3);
    if (residual) {
#pragma unroll
        for (int i = 0; i < 4; i++)
#pragma unroll
            for (int j = 0; j < 2; j++)
#pragma unroll
                for (int v = 0; v < 8; v++) {
                    long idx = (long)(mb + 16 * i + v) * N + (nc + 16 * j);
                    C[idx] = acc[i][j][v] + residual[idx];
                }
    } else {
#pragma unroll
        for (int i = 0; i < 4; i++)
#pragma unroll
            for (int j = 0; j < 2; j++)
#pragma unroll
                for (int v = 0; v < 8; v++) {
                    long idx = (long)(mb + 16 * i + v) * N + (nc + 16 * j);
                    C[idx] = acc[i][j][v];
                }
    }
}

// ---------------------------------------------------------------------------
// f32 -> bf16 weight conversion
// ---------------------------------------------------------------------------
__global__ void cvt_bf16_kernel(const float* __restrict__ src, u16* __restrict__ dst, long n) {
    long i = (long)blockIdx.x * blockDim.x + threadIdx.x;
    if (i < n) dst[i] = f32_to_bf16(src[i]);
}

// ---------------------------------------------------------------------------
// gemma_norm: out_bf16[row] = rms(x[row]) * (1 + w)
// ---------------------------------------------------------------------------
__global__ void __launch_bounds__(256) rmsnorm_bf16_kernel(
    const float* __restrict__ x, const float* __restrict__ w,
    u16* __restrict__ out, int D)
{
    int row = blockIdx.x;
    const float* xr = x + (long)row * D;
    __shared__ float red[8];
    float s = 0.f;
    for (int i = threadIdx.x; i < D; i += 256) { float v = xr[i]; s += v * v; }
    for (int o = 16; o > 0; o >>= 1) s += __shfl_xor(s, o, 32);
    if ((threadIdx.x & 31) == 0) red[threadIdx.x >> 5] = s;
    __syncthreads();
    if (threadIdx.x < 8) {
        float t = red[threadIdx.x];
        for (int o = 4; o > 0; o >>= 1) t += __shfl_xor(t, o, 8);
        if (threadIdx.x == 0) red[0] = t;
    }
    __syncthreads();
    float inv = rsqrtf(red[0] / (float)D + 1e-6f);
    for (int i = threadIdx.x; i < D; i += 256)
        out[(long)row * D + i] = f32_to_bf16(xr[i] * inv * (1.0f + w[i]));
}

// ---------------------------------------------------------------------------
// Attention head prep: per-head gemma norm + partial RoPE, repack to bf16.
// ---------------------------------------------------------------------------
__global__ void __launch_bounds__(128) attn_prep_kernel(
    const float* __restrict__ qg, const float* __restrict__ katt,
    const float* __restrict__ vatt, const float* __restrict__ qn_w,
    const float* __restrict__ kn_w, const int* __restrict__ pos,
    u16* __restrict__ Qh, u16* __restrict__ Kh, u16* __restrict__ Vt, int T)
{
    int t = blockIdx.x;
    int h = blockIdx.y;
    int d = threadIdx.x;

    float val;
    const float* nw;
    if (h < 16) { val = qg[((long)t * 16 + h) * 256 + d];        nw = qn_w; }
    else        { val = katt[((long)t * 4 + (h - 16)) * 128 + d]; nw = kn_w; }

    float s = val * val;
    for (int o = 16; o > 0; o >>= 1) s += __shfl_xor(s, o, 32);
    __shared__ float red[4];
    if ((d & 31) == 0) red[d >> 5] = s;
    __syncthreads();
    float tot = red[0] + red[1] + red[2] + red[3];
    float v = val * rsqrtf(tot / 128.f + 1e-6f) * (1.f + nw[d]);

    __shared__ float vbuf[128];
    vbuf[d] = v;
    __syncthreads();

    float fp = (float)pos[t];
    float outv = v;
    if (d < 16) {
        float fr = fp * __powf(10000.f, -(2.f * d) / 32.f);
        outv = vbuf[d] * __cosf(fr) - vbuf[d + 16] * __sinf(fr);
    } else if (d < 32) {
        int i = d - 16;
        float fr = fp * __powf(10000.f, -(2.f * i) / 32.f);
        outv = vbuf[d] * __cosf(fr) + vbuf[i] * __sinf(fr);
    }

    if (h < 16) {
        Qh[((long)h * T + t) * 128 + d] = f32_to_bf16(outv);
    } else {
        int kh = h - 16;
        Kh[((long)kh * T + t) * 128 + d] = f32_to_bf16(outv);
        Vt[((long)kh * 128 + d) * T + t] =
            f32_to_bf16(vatt[((long)t * 4 + kh) * 128 + d]);
    }
}

// ---------------------------------------------------------------------------
// Flash attention with WMMA + TDM staging.
//   Block = 4 waves, each wave owns 16 queries of a 64-query tile.
//   Wave 0 DMAs the 32-key K tile (32x128) and V tile (128x32, hd-major)
//   into LDS via tensor_load_to_lds, gated by s_wait_tensorcnt + barrier.
//   S = Q*K^T per 16x16 tile; P transposed through LDS; O += P*V.
// ---------------------------------------------------------------------------
__global__ void __launch_bounds__(128) flash_attn_kernel(
    const u16* __restrict__ Qh, const u16* __restrict__ Kh,
    const u16* __restrict__ Vt, const float* __restrict__ qg,
    const int* __restrict__ pos, u16* __restrict__ Ybf, int T)
{
    int h    = blockIdx.y;
    int kvh  = h >> 2;
    int lane = threadIdx.x & 31;
    int wave = threadIdx.x >> 5;
    int q0   = blockIdx.x * 64 + wave * 16;

    const u16* Q  = Qh + (long)h * T * 128;
    const u16* Kp = Kh + (long)kvh * T * 128;
    const u16* Vp = Vt + (long)kvh * 128 * T;

    v16bf qf[4];
#pragma unroll
    for (int c = 0; c < 4; c++) qf[c] = load_frag_rowmajor(Q, 128, q0, c * 32, lane);

    v8f oacc[8];
#pragma unroll
    for (int c = 0; c < 8; c++)
#pragma unroll
        for (int v = 0; v < 8; v++) oacc[c][v] = 0.f;
    float rowmax[8], rowsum[8];
#pragma unroll
    for (int v = 0; v < 8; v++) { rowmax[v] = -1e30f; rowsum[v] = 0.f; }

    __shared__ __align__(16) u16 plds[4][16][32];
    __shared__ __align__(16) u16 ktile[32][128];   // [key][hd]
    __shared__ __align__(16) u16 vtile[128][32];   // [hd][key]
    unsigned koff = (unsigned)(unsigned long long)(void*)&ktile[0][0];
    unsigned voff = (unsigned)(unsigned long long)(void*)&vtile[0][0];

    const float scale = 0.08838834764831845f;  // 1/sqrt(128)
    int nchunks = (blockIdx.x + 1) * 2;        // 32 keys per chunk, causal bound

    int ncol = lane & 15;
    int mrow = (lane >> 4) * 8;

    for (int ch = 0; ch < nchunks; ch++) {
        int kbase = ch * 32;
        if (wave == 0) {
            // K rows [kbase, kbase+32), full 128-wide rows, stride 128 elems
            tdm_load_2d_bf16(koff, Kp + (long)kbase * 128, 128u, 32u, 128ull);
            // V tile: 128 hd rows x 32 keys, stride T elems
            tdm_load_2d_bf16(voff, Vp + kbase, 32u, 128u, (unsigned long long)T);
            __builtin_amdgcn_s_wait_tensorcnt(0);
        }
        __syncthreads();   // K/V tiles visible to all waves

        v8f sc[2];
#pragma unroll
        for (int v = 0; v < 8; v++) { sc[0][v] = 0.f; sc[1][v] = 0.f; }
#pragma unroll
        for (int c = 0; c < 4; c++) {
            v16bf kf0 = load_frag_rowmajor(&ktile[0][0], 128, 0,  c * 32, lane);
            v16bf kf1 = load_frag_rowmajor(&ktile[0][0], 128, 16, c * 32, lane);
            sc[0] = __builtin_amdgcn_wmma_f32_16x16x32_bf16(false, qf[c], false, kf0,
                                                            (short)0, sc[0], false, false);
            sc[1] = __builtin_amdgcn_wmma_f32_16x16x32_bf16(false, qf[c], false, kf1,
                                                            (short)0, sc[1], false, false);
        }
#pragma unroll
        for (int v = 0; v < 8; v++) {
            int p = pos[q0 + mrow + v];
            float s0 = sc[0][v] * scale;
            float s1 = sc[1][v] * scale;
            if (kbase + ncol > p)      s0 = -1e30f;
            if (kbase + 16 + ncol > p) s1 = -1e30f;
            float mv = fmaxf(s0, s1);
#pragma unroll
            for (int o = 1; o < 16; o <<= 1) mv = fmaxf(mv, __shfl_xor(mv, o, 32));
            float nmax = fmaxf(rowmax[v], mv);
            float corr = __expf(rowmax[v] - nmax);
            rowsum[v] *= corr;
#pragma unroll
            for (int c = 0; c < 8; c++) oacc[c][v] *= corr;
            float p0 = __expf(s0 - nmax);
            float p1 = __expf(s1 - nmax);
            float ps = p0 + p1;
#pragma unroll
            for (int o = 1; o < 16; o <<= 1) ps += __shfl_xor(ps, o, 32);
            rowsum[v] += ps;
            rowmax[v]  = nmax;
            plds[wave][mrow + v][ncol]      = f32_to_bf16(p0);
            plds[wave][mrow + v][16 + ncol] = f32_to_bf16(p1);
        }
        __syncthreads();   // P tiles ready
        v16bf pf = load_frag_rowmajor(&plds[wave][0][0], 32, 0, 0, lane);
#pragma unroll
        for (int c = 0; c < 8; c++) {
            v16bf vf = load_frag_rowmajor(&vtile[0][0], 32, c * 16, 0, lane);
            oacc[c] = __builtin_amdgcn_wmma_f32_16x16x32_bf16(false, pf, false, vf,
                                                              (short)0, oacc[c], false, false);
        }
        __syncthreads();   // done with K/V/P tiles before next DMA overwrites
    }

#pragma unroll
    for (int c = 0; c < 8; c++)
#pragma unroll
        for (int v = 0; v < 8; v++) {
            int t  = q0 + mrow + v;
            int hd = c * 16 + ncol;
            float o = oacc[c][v] / rowsum[v];
            float gate = qg[((long)t * 16 + h) * 256 + 128 + hd];
            float sg = 1.f / (1.f + __expf(-gate));
            Ybf[(long)t * 2048 + h * 128 + hd] = f32_to_bf16(o * sg);
        }
}

// ---------------------------------------------------------------------------
// Depthwise causal conv (K=4) + SiLU over the DeltaNet mixed projection.
// ---------------------------------------------------------------------------
__global__ void conv_silu_kernel(const float* __restrict__ mixed,
                                 const float* __restrict__ cw,
                                 float* __restrict__ qkv, long total)
{
    long i = (long)blockIdx.x * blockDim.x + threadIdx.x;
    if (i >= total) return;
    int t = (int)(i >> 13);        // / 8192
    int c = (int)(i & 8191);
    float acc = 0.f;
#pragma unroll
    for (int j = 0; j < 4; j++) {
        int tt = t - 3 + j;
        if (tt >= 0) acc += mixed[(long)tt * 8192 + c] * cw[c * 4 + j];
    }
    qkv[i] = acc / (1.f + __expf(-acc));
}

// ---------------------------------------------------------------------------
// L2-normalize q (heads 0..15) and k (heads 16..31) 128-dim vectors in place.
// ---------------------------------------------------------------------------
__global__ void __launch_bounds__(128) l2norm_kernel(float* __restrict__ qkv)
{
    int t = blockIdx.x, h = blockIdx.y, d = threadIdx.x;
    long off = (long)t * 8192 + (h < 16 ? h * 128 : 2048 + (h - 16) * 128);
    float v = qkv[off + d];
    float s = v * v;
    for (int o = 16; o > 0; o >>= 1) s += __shfl_xor(s, o, 32);
    __shared__ float red[4];
    if ((d & 31) == 0) red[d >> 5] = s;
    __syncthreads();
    float tot = red[0] + red[1] + red[2] + red[3];
    qkv[off + d] = v * (1.f / fmaxf(sqrtf(tot), 1e-12f));
}

// ---------------------------------------------------------------------------
// beta = sigmoid(bb); g = -exp(A_log) * softplus(aa + dt_bias)
// ---------------------------------------------------------------------------
__global__ void betag_kernel(const float* __restrict__ bb, const float* __restrict__ aa,
                             const float* __restrict__ A_log, const float* __restrict__ dt_bias,
                             float* __restrict__ beta, float* __restrict__ g, int n)
{
    int i = blockIdx.x * blockDim.x + threadIdx.x;
    if (i >= n) return;
    int h = i & 31;
    beta[i] = 1.f / (1.f + __expf(-bb[i]));
    float x = aa[i] + dt_bias[h];
    float sp = (x > 20.f) ? x : log1pf(__expf(x));
    g[i] = -__expf(A_log[h]) * sp;
}

// ---------------------------------------------------------------------------
// Gated-delta recurrent scan.  One block per value head (32); thread d owns
// column dv=d of the 128x128 state S (kept in VGPRs).  q/k broadcast via LDS.
// ---------------------------------------------------------------------------
__global__ void __launch_bounds__(128) deltanet_scan_kernel(
    const float* __restrict__ qkv, const float* __restrict__ beta,
    const float* __restrict__ g, float* __restrict__ o, int T)
{
    int hv = blockIdx.x;
    int hk = hv >> 1;
    int dv = threadIdx.x;
    float S[128];
#pragma unroll
    for (int i = 0; i < 128; i++) S[i] = 0.f;
    __shared__ float kq[256];
    const float scale = 0.08838834764831845f;  // DK^-0.5

    for (int t = 0; t < T; t++) {
        long base = (long)t * 8192;
        kq[dv]       = qkv[base + 2048 + hk * 128 + dv];  // k
        kq[128 + dv] = qkv[base + hk * 128 + dv];          // q
        __syncthreads();
        float eg = __expf(g[t * 32 + hv]);
        float bt = beta[t * 32 + hv];
        float vt = qkv[base + 4096 + hv * 128 + dv];
        float dot = 0.f;
#pragma unroll
        for (int i = 0; i < 128; i++) dot += kq[i] * S[i];
        float delta = bt * (vt - eg * dot);
        float ov = 0.f;
#pragma unroll
        for (int i = 0; i < 128; i++) {
            float s = S[i] * eg + kq[i] * delta;
            S[i] = s;
            ov += kq[128 + i] * s;
        }
        o[((long)t * 32 + hv) * 128 + dv] = ov * scale;
        __syncthreads();
    }
}

// ---------------------------------------------------------------------------
// DeltaNet output gating: rms(o) * norm_w * silu(z) -> bf16
// ---------------------------------------------------------------------------
__global__ void __launch_bounds__(128) dn_gate_kernel(
    const float* __restrict__ o, const float* __restrict__ z,
    const float* __restrict__ norm_w, u16* __restrict__ ozbf)
{
    int t = blockIdx.x, hv = blockIdx.y, d = threadIdx.x;
    long oi = ((long)t * 32 + hv) * 128 + d;
    float v = o[oi];
    float s = v * v;
    for (int off = 16; off > 0; off >>= 1) s += __shfl_xor(s, off, 32);
    __shared__ float red[4];
    if ((d & 31) == 0) red[d >> 5] = s;
    __syncthreads();
    float tot = red[0] + red[1] + red[2] + red[3];
    float inv = rsqrtf(tot / 128.f + 1e-6f);
    float zz = z[(long)t * 4096 + hv * 128 + d];
    float sil = zz / (1.f + __expf(-zz));
    ozbf[(long)t * 4096 + hv * 128 + d] = f32_to_bf16(v * inv * norm_w[d] * sil);
}

// ---------------------------------------------------------------------------
// Host side
// ---------------------------------------------------------------------------
static inline char* ws_alloc(char*& p, size_t bytes) {
    char* r = p;
    p += (bytes + 255) & ~(size_t)255;
    return r;
}

extern "C" void kernel_launch(void* const* d_in, const int* in_sizes, int n_in,
                              void* d_out, int out_size, void* d_ws, size_t ws_size,
                              hipStream_t stream)
{
    const float* x       = (const float*)d_in[0];
    const int*   pos     = (const int*)  d_in[1];
    const float* ln1_w   = (const float*)d_in[2];
    const float* ln2_w   = (const float*)d_in[3];
    const float* q_w     = (const float*)d_in[4];
    const float* k_w     = (const float*)d_in[5];
    const float* v_w     = (const float*)d_in[6];
    const float* o_w     = (const float*)d_in[7];
    const float* qn_w    = (const float*)d_in[8];
    const float* kn_w    = (const float*)d_in[9];
    const float* qkv_w   = (const float*)d_in[10];
    const float* z_w     = (const float*)d_in[11];
    const float* b_w     = (const float*)d_in[12];
    const float* a_w     = (const float*)d_in[13];
    const float* conv_w  = (const float*)d_in[14];
    const float* dt_bias = (const float*)d_in[15];
    const float* A_log   = (const float*)d_in[16];
    const float* norm_w  = (const float*)d_in[17];
    const float* out_w   = (const float*)d_in[18];

    const int T = T_SEQ, D = D_MODEL;

    char* p = (char*)d_ws;
    u16*  qw_bf   = (u16*)ws_alloc(p, (size_t)4096 * 2048 * 2);
    u16*  kw_bf   = (u16*)ws_alloc(p, (size_t)512  * 2048 * 2);
    u16*  vw_bf   = (u16*)ws_alloc(p, (size_t)512  * 2048 * 2);
    u16*  ow_bf   = (u16*)ws_alloc(p, (size_t)2048 * 2048 * 2);
    u16*  qkvw_bf = (u16*)ws_alloc(p, (size_t)8192 * 2048 * 2);
    u16*  zw_bf   = (u16*)ws_alloc(p, (size_t)4096 * 2048 * 2);
    u16*  bw_bf   = (u16*)ws_alloc(p, (size_t)32   * 2048 * 2);
    u16*  aw_bf   = (u16*)ws_alloc(p, (size_t)32   * 2048 * 2);
    u16*  outw_bf = (u16*)ws_alloc(p, (size_t)2048 * 4096 * 2);
    u16*  xn1     = (u16*)ws_alloc(p, (size_t)T * D * 2);
    u16*  xn2     = (u16*)ws_alloc(p, (size_t)T * D * 2);
    float* qg     = (float*)ws_alloc(p, (size_t)T * 4096 * 4);
    float* katt   = (float*)ws_alloc(p, (size_t)T * 512 * 4);
    float* vatt   = (float*)ws_alloc(p, (size_t)T * 512 * 4);
    u16*  Qh      = (u16*)ws_alloc(p, (size_t)16 * T * 128 * 2);
    u16*  Kh      = (u16*)ws_alloc(p, (size_t)4  * T * 128 * 2);
    u16*  Vt      = (u16*)ws_alloc(p, (size_t)4  * 128 * T * 2);
    u16*  ybf     = (u16*)ws_alloc(p, (size_t)T * 2048 * 2);
    float* hbuf   = (float*)ws_alloc(p, (size_t)T * D * 4);
    float* mixed  = (float*)ws_alloc(p, (size_t)T * 8192 * 4);
    float* zbuf   = (float*)ws_alloc(p, (size_t)T * 4096 * 4);
    float* bb     = (float*)ws_alloc(p, (size_t)T * 32 * 4);
    float* aa     = (float*)ws_alloc(p, (size_t)T * 32 * 4);
    float* qkv    = (float*)ws_alloc(p, (size_t)T * 8192 * 4);
    float* betab  = (float*)ws_alloc(p, (size_t)T * 32 * 4);
    float* gbuf   = (float*)ws_alloc(p, (size_t)T * 32 * 4);
    float* odn    = (float*)ws_alloc(p, (size_t)T * 32 * 128 * 4);
    u16*  ozbf    = (u16*)ws_alloc(p, (size_t)T * 4096 * 2);

    auto cvt = [&](const float* s, u16* dst, long n) {
        cvt_bf16_kernel<<<(int)((n + 255) / 256), 256, 0, stream>>>(s, dst, n);
    };
    auto gemm = [&](const u16* A, const u16* W, float* C, const float* R,
                    int M, int N, int K) {
        int tiles = (M / 64) * (N / 32);
        wmma_gemm_kernel<<<(tiles + 7) / 8, 256, 0, stream>>>(A, W, C, R, M, N, K);
    };

    // 1) weights -> bf16
    cvt(q_w,   qw_bf,   (long)4096 * 2048);
    cvt(k_w,   kw_bf,   (long)512  * 2048);
    cvt(v_w,   vw_bf,   (long)512  * 2048);
    cvt(o_w,   ow_bf,   (long)2048 * 2048);
    cvt(qkv_w, qkvw_bf, (long)8192 * 2048);
    cvt(z_w,   zw_bf,   (long)4096 * 2048);
    cvt(b_w,   bw_bf,   (long)32   * 2048);
    cvt(a_w,   aw_bf,   (long)32   * 2048);
    cvt(out_w, outw_bf, (long)2048 * 4096);

    // 2) attention branch
    rmsnorm_bf16_kernel<<<T, 256, 0, stream>>>(x, ln1_w, xn1, D);
    gemm(xn1, qw_bf, qg,   nullptr, T, 4096, D);
    gemm(xn1, kw_bf, katt, nullptr, T, 512,  D);
    gemm(xn1, vw_bf, vatt, nullptr, T, 512,  D);
    attn_prep_kernel<<<dim3(T, 20), 128, 0, stream>>>(qg, katt, vatt, qn_w, kn_w,
                                                      pos, Qh, Kh, Vt, T);
    flash_attn_kernel<<<dim3(T / 64, 16), 128, 0, stream>>>(Qh, Kh, Vt, qg, pos, ybf, T);
    gemm(ybf, ow_bf, hbuf, x, T, D, 2048);   // h = x + attn(...)

    // 3) deltanet branch
    rmsnorm_bf16_kernel<<<T, 256, 0, stream>>>(hbuf, ln2_w, xn2, D);
    gemm(xn2, qkvw_bf, mixed, nullptr, T, 8192, D);
    gemm(xn2, zw_bf,   zbuf,  nullptr, T, 4096, D);
    gemm(xn2, bw_bf,   bb,    nullptr, T, 32,   D);
    gemm(xn2, aw_bf,   aa,    nullptr, T, 32,   D);
    {
        long total = (long)T * 8192;
        conv_silu_kernel<<<(int)((total + 255) / 256), 256, 0, stream>>>(mixed, conv_w,
                                                                         qkv, total);
    }
    l2norm_kernel<<<dim3(T, 32), 128, 0, stream>>>(qkv);
    betag_kernel<<<(T * 32 + 255) / 256, 256, 0, stream>>>(bb, aa, A_log, dt_bias,
                                                           betab, gbuf, T * 32);
    deltanet_scan_kernel<<<32, 128, 0, stream>>>(qkv, betab, gbuf, odn, T);
    dn_gate_kernel<<<dim3(T, 32), 128, 0, stream>>>(odn, zbuf, norm_w, ozbf);
    gemm(ozbf, outw_bf, (float*)d_out, hbuf, T, D, 4096);  // out = h + deltanet(...)

    (void)in_sizes; (void)n_in; (void)out_size; (void)ws_size;
}
